// CognitiveAttention_47227460386902
// MI455X (gfx1250) — compile-verified, matrix operand order checked by reading, not run on hardware
//
#include <hip/hip_runtime.h>
#include <hip/hip_bf16.h>

// ---------------------------------------------------------------------------
// Problem constants (from reference): B=4, S=2048, H=1024, NH=16, HD=64
// ---------------------------------------------------------------------------
#define B_  4
#define S_  2048
#define H_  1024
#define NH_ 16
#define HD_ 64
#define NT_ (B_ * S_)          // 8192 tokens

typedef __bf16 bf16_t;
typedef __attribute__((ext_vector_type(16))) __bf16 v16bf;
typedef __attribute__((ext_vector_type(8)))  __bf16 v8bf;
typedef __attribute__((ext_vector_type(8)))  float  v8f;

__device__ __forceinline__ v16bf make_frag(const bf16_t* lo, const bf16_t* hi) {
    union { v16bf v; v8bf h[2]; } u;
    u.h[0] = *(const v8bf*)lo;
    u.h[1] = *(const v8bf*)hi;
    return u.v;
}

__device__ __forceinline__ v8f wmma_bf16(v16bf a, v16bf b, v8f c) {
    // D = A(16x32 bf16) x B(32x16 bf16) + C(16x16 f32)
    return __builtin_amdgcn_wmma_f32_16x16x32_bf16(
        /*neg_a=*/false, a, /*neg_b=*/false, b,
        /*c_mod=*/(short)0, c, /*reuse_a=*/false, /*reuse_b=*/false);
}

// CDNA5 async global->LDS copy, 16 bytes per lane, tracked with ASYNCcnt.
__device__ __forceinline__ void async_ld_b128(unsigned lds_byte_off,
                                              const bf16_t* gaddr) {
    asm volatile("global_load_async_to_lds_b128 %0, %1, off"
                 :: "v"(lds_byte_off), "v"(gaddr) : "memory");
}
#define WAIT_ASYNC_6() asm volatile("s_wait_asynccnt 0x6" ::: "memory")
#define WAIT_ASYNC_0() asm volatile("s_wait_asynccnt 0x0" ::: "memory")

// ---------------------------------------------------------------------------
// f32 -> bf16 conversion
// ---------------------------------------------------------------------------
__global__ void cvt_f32_bf16(const float* __restrict__ src,
                             bf16_t* __restrict__ dst, int n) {
    int idx = blockIdx.x * blockDim.x + threadIdx.x;
    if (idx < n) dst[idx] = (bf16_t)src[idx];
}

// ---------------------------------------------------------------------------
// Tiled WMMA GEMM:  Y[M,N] = A[M,K] * W[N,K]^T   (A, W bf16; accumulate f32)
// Block = 128 threads (4 waves). Tile 128x64, BK=32, double-buffered LDS
// filled with async-to-LDS copies. Wave owns a 32x64 strip (8 accumulators).
// mode 0: store bf16 permuted [b][nh][s][hd]   (Q/K projections)
// mode 2: store bf16 permuted [b][nh][hd][s]   (V, pre-transposed per head)
// mode 1: store f32 row-major [M,N]            (final output projection)
// ---------------------------------------------------------------------------
__global__ __launch_bounds__(128)
void gemm_bf16_kernel(const bf16_t* __restrict__ A,
                      const bf16_t* __restrict__ W,
                      void* __restrict__ C,
                      int M, int N, int K, int mode) {
    __shared__ __align__(16) bf16_t As[2][128][32];
    __shared__ __align__(16) bf16_t Bs[2][64][32];

    const int tid  = threadIdx.x;
    const int wave = tid >> 5;
    const int lane = tid & 31;
    const int g    = lane >> 4;      // K-half selector
    const int ln   = lane & 15;      // row (A) / col (B,C) within 16
    const int tile_n = blockIdx.x * 64;
    const int tile_m = blockIdx.y * 128;

    const unsigned asbase = (unsigned)(unsigned long long)&As[0][0][0];
    const unsigned bsbase = (unsigned)(unsigned long long)&Bs[0][0][0];

    v8f acc[2][4];
    for (int rr = 0; rr < 2; rr++)
        for (int c = 0; c < 4; c++)
            for (int j = 0; j < 8; j++) acc[rr][c][j] = 0.0f;

    // Stage one 128x32 A tile + 64x32 B tile: 768 16B chunks / 128 thr = 6.
    auto stage = [&](int buf, int k0) {
        unsigned la = asbase + (unsigned)(buf * 128 * 32 * 2);
        unsigned lb = bsbase + (unsigned)(buf * 64 * 32 * 2);
        #pragma unroll
        for (int rep = 0; rep < 4; rep++) {
            int ci = tid + 128 * rep;
            int r = ci >> 2, kc = (ci & 3) * 8;
            async_ld_b128(la + (unsigned)(r * 64 + kc * 2),
                          A + (size_t)(tile_m + r) * K + k0 + kc);
        }
        #pragma unroll
        for (int rep = 0; rep < 2; rep++) {
            int ci = tid + 128 * rep;
            int r = ci >> 2, kc = (ci & 3) * 8;
            async_ld_b128(lb + (unsigned)(r * 64 + kc * 2),
                          W + (size_t)(tile_n + r) * K + k0 + kc);
        }
    };

    const int KT = K >> 5;           // K/32 steps
    stage(0, 0);
    for (int kt = 0; kt < KT; kt++) {
        const int cur = kt & 1;
        if (kt + 1 < KT) {
            stage(cur ^ 1, (kt + 1) * 32);
            WAIT_ASYNC_6();          // previous tile's 6 chunks landed
        } else {
            WAIT_ASYNC_0();
        }
        __syncthreads();

        v16bf af0 = make_frag(&As[cur][wave * 32 + ln][g * 8],
                              &As[cur][wave * 32 + ln][16 + g * 8]);
        v16bf af1 = make_frag(&As[cur][wave * 32 + 16 + ln][g * 8],
                              &As[cur][wave * 32 + 16 + ln][16 + g * 8]);
        #pragma unroll
        for (int c = 0; c < 4; c++) {
            v16bf bf = make_frag(&Bs[cur][c * 16 + ln][g * 8],
                                 &Bs[cur][c * 16 + ln][16 + g * 8]);
            acc[0][c] = wmma_bf16(af0, bf, acc[0][c]);
            acc[1][c] = wmma_bf16(af1, bf, acc[1][c]);
        }
        __syncthreads();             // reads done before buffer reuse
    }

    #pragma unroll
    for (int rr = 0; rr < 2; rr++) {
        if (mode == 0) {
            bf16_t* Cb = (bf16_t*)C;
            #pragma unroll
            for (int c = 0; c < 4; c++)
                #pragma unroll
                for (int j = 0; j < 8; j++) {
                    int m = tile_m + wave * 32 + rr * 16 + j + 8 * g;
                    int n = tile_n + c * 16 + ln;
                    int b  = m >> 11, s = m & (S_ - 1);
                    int nh = n >> 6,  hd = n & (HD_ - 1);
                    Cb[(((size_t)(b * NH_ + nh)) * S_ + s) * HD_ + hd] =
                        (bf16_t)acc[rr][c][j];
                }
        } else if (mode == 2) {
            bf16_t* Cb = (bf16_t*)C;
            #pragma unroll
            for (int c = 0; c < 4; c++)
                #pragma unroll
                for (int j = 0; j < 8; j++) {
                    int m = tile_m + wave * 32 + rr * 16 + j + 8 * g;
                    int n = tile_n + c * 16 + ln;
                    int b  = m >> 11, s = m & (S_ - 1);
                    int nh = n >> 6,  hd = n & (HD_ - 1);
                    Cb[(((size_t)(b * NH_ + nh)) * HD_ + hd) * S_ + s] =
                        (bf16_t)acc[rr][c][j];
                }
        } else {
            float* Cf = (float*)C;
            #pragma unroll
            for (int c = 0; c < 4; c++)
                #pragma unroll
                for (int j = 0; j < 8; j++) {
                    int m = tile_m + wave * 32 + rr * 16 + j + 8 * g;
                    int n = tile_n + c * 16 + ln;
                    Cf[(size_t)m * N + n] = acc[rr][c][j];
                }
        }
    }
}

// ---------------------------------------------------------------------------
// RoPE (in place on bf16 [B][NH][S][HD]); thread handles pair (i, i+32).
// ---------------------------------------------------------------------------
__global__ void rope_kernel(bf16_t* __restrict__ buf, int total) {
    int idx = blockIdx.x * blockDim.x + threadIdx.x;
    if (idx >= total) return;
    int i = idx & 31;
    int r = idx >> 5;                // row over B*NH*S
    int spos = r & (S_ - 1);
    size_t base = (size_t)r * HD_;
    float ang = (float)spos * __expf(-0.28782313662425572f * (float)i);
    float sn, cs;
    __sincosf(ang, &sn, &cs);
    float x0 = (float)buf[base + i];
    float x1 = (float)buf[base + 32 + i];
    buf[base + i]      = (bf16_t)(x0 * cs - x1 * sn);
    buf[base + 32 + i] = (bf16_t)(x1 * cs + x0 * sn);
}

// ---------------------------------------------------------------------------
// Flash attention: one wave per (b, h, 16-row q tile). V is pre-transposed
// per head ([b][h][HD][S]) so P x V B-fragments load straight from global.
// Only P round-trips through LDS (f32 C-layout -> bf16 A-fragment).
// ---------------------------------------------------------------------------
__global__ __launch_bounds__(32)
void attn_kernel(const bf16_t* __restrict__ Q,
                 const bf16_t* __restrict__ Kmat,
                 const bf16_t* __restrict__ VT,
                 bf16_t* __restrict__ O) {
    __shared__ __align__(16) bf16_t Ps[16][32];   // probs tile (q x key)

    const int lane = threadIdx.x;
    const int g  = lane >> 4;
    const int ln = lane & 15;

    const int idx = blockIdx.x;
    const int qt = idx & 127;            // S/16 = 128 q tiles
    const int h  = (idx >> 7) & (NH_ - 1);
    const int b  = idx >> 11;

    const size_t headoff = ((size_t)(b * NH_ + h)) * S_ * HD_;
    const bf16_t* Qp = Q + headoff;
    const bf16_t* Kp = Kmat + headoff;
    const bf16_t* Vp = VT + headoff;     // [HD][S] within head
    const int m0 = qt * 16;

    // Q fragments: 16 rows x 64 dims -> two K=32 A-frags
    v16bf aq[2];
    #pragma unroll
    for (int kk = 0; kk < 2; kk++) {
        const bf16_t* qrow = Qp + (size_t)(m0 + ln) * HD_ + kk * 32;
        aq[kk] = make_frag(qrow + g * 8, qrow + 16 + g * 8);
    }

    float mrow[8], lrow[8];
    v8f oacc[4];
    #pragma unroll
    for (int j = 0; j < 8; j++) { mrow[j] = -3.0e38f; lrow[j] = 0.0f; }
    for (int c = 0; c < 4; c++)
        for (int j = 0; j < 8; j++) oacc[c][j] = 0.0f;

    const float scale = 0.125f;          // 1/sqrt(64)

    for (int kt = 0; kt < S_ / 32; kt++) {
        // ---- scores: S = Q x K^T for 32 keys (two 16-col groups) ----
        v8f sc0, sc1;
        for (int j = 0; j < 8; j++) { sc0[j] = 0.0f; sc1[j] = 0.0f; }
        const bf16_t* kr0 = Kp + (size_t)(kt * 32 + ln) * HD_;
        const bf16_t* kr1 = Kp + (size_t)(kt * 32 + 16 + ln) * HD_;
        #pragma unroll
        for (int kk = 0; kk < 2; kk++) {
            v16bf b0 = make_frag(kr0 + kk * 32 + g * 8,
                                 kr0 + kk * 32 + 16 + g * 8);
            sc0 = wmma_bf16(aq[kk], b0, sc0);
            v16bf b1 = make_frag(kr1 + kk * 32 + g * 8,
                                 kr1 + kk * 32 + 16 + g * 8);
            sc1 = wmma_bf16(aq[kk], b1, sc1);
        }

        // ---- online softmax over this 32-key tile ----
        #pragma unroll
        for (int j = 0; j < 8; j++) {
            float s0 = sc0[j] * scale;
            float s1 = sc1[j] * scale;
            float mx = fmaxf(s0, s1);
            for (int msk = 1; msk < 16; msk <<= 1)
                mx = fmaxf(mx, __shfl_xor(mx, msk, 32));
            float mnew = fmaxf(mrow[j], mx);
            float p0 = __expf(s0 - mnew);
            float p1 = __expf(s1 - mnew);
            float rs = p0 + p1;
            for (int msk = 1; msk < 16; msk <<= 1)
                rs += __shfl_xor(rs, msk, 32);
            float corr = __expf(mrow[j] - mnew);
            lrow[j] = lrow[j] * corr + rs;
            mrow[j] = mnew;
            #pragma unroll
            for (int c = 0; c < 4; c++) oacc[c][j] *= corr;
            // C-layout (row j+8g, col ln / 16+ln) -> LDS for A-frag reload
            Ps[j + 8 * g][ln]      = (bf16_t)p0;
            Ps[j + 8 * g][16 + ln] = (bf16_t)p1;
        }
        __syncthreads();

        // ---- O += P x V  (B-frags straight from pre-transposed V) ----
        v16bf ap = make_frag(&Ps[ln][g * 8], &Ps[ln][16 + g * 8]);
        #pragma unroll
        for (int c = 0; c < 4; c++) {
            const bf16_t* vrow = Vp + (size_t)(c * 16 + ln) * S_ + kt * 32;
            v16bf bv = make_frag(vrow + g * 8, vrow + 16 + g * 8);
            oacc[c] = wmma_bf16(ap, bv, oacc[c]);
        }
        __syncthreads();
    }

    // ---- epilogue: normalize, store merged-head bf16 [B][S][NH][HD] ----
    #pragma unroll
    for (int j = 0; j < 8; j++) {
        float inv = 1.0f / lrow[j];
        int spos = m0 + j + 8 * g;
        #pragma unroll
        for (int c = 0; c < 4; c++) {
            int hd = c * 16 + ln;
            O[(((size_t)(b * S_ + spos)) * NH_ + h) * HD_ + hd] =
                (bf16_t)(oacc[c][j] * inv);
        }
    }
}

// ---------------------------------------------------------------------------
// Host-side orchestration
// ---------------------------------------------------------------------------
extern "C" void kernel_launch(void* const* d_in, const int* in_sizes, int n_in,
                              void* d_out, int out_size, void* d_ws, size_t ws_size,
                              hipStream_t stream) {
    const float* X  = (const float*)d_in[0];
    const float* Wq = (const float*)d_in[1];
    const float* Wk = (const float*)d_in[2];
    const float* Wv = (const float*)d_in[3];
    const float* Wo = (const float*)d_in[4];
    float* out = (float*)d_out;

    char* ws = (char*)d_ws;
    size_t off = 0;
    auto alloc = [&](size_t bytes) -> char* {
        char* p = ws + off;
        off += (bytes + 255) & ~(size_t)255;
        return p;
    };

    const size_t xe = (size_t)NT_ * H_;       // 8,388,608 elements
    const size_t we = (size_t)H_ * H_;        // 1,048,576 elements

    bf16_t* Xb  = (bf16_t*)alloc(xe * 2);
    bf16_t* Wqb = (bf16_t*)alloc(we * 2);
    bf16_t* Wkb = (bf16_t*)alloc(we * 2);
    bf16_t* Wvb = (bf16_t*)alloc(we * 2);
    bf16_t* Wob = (bf16_t*)alloc(we * 2);
    bf16_t* Qb  = (bf16_t*)alloc(xe * 2);
    bf16_t* Kb  = (bf16_t*)alloc(xe * 2);
    bf16_t* Vb  = (bf16_t*)alloc(xe * 2);     // transposed per head [b][h][hd][s]
    bf16_t* Ab  = (bf16_t*)alloc(xe * 2);     // merged-head attention output

    // 1) convert inputs to bf16
    cvt_f32_bf16<<<(int)((xe + 255) / 256), 256, 0, stream>>>(X,  Xb,  (int)xe);
    cvt_f32_bf16<<<(int)((we + 255) / 256), 256, 0, stream>>>(Wq, Wqb, (int)we);
    cvt_f32_bf16<<<(int)((we + 255) / 256), 256, 0, stream>>>(Wk, Wkb, (int)we);
    cvt_f32_bf16<<<(int)((we + 255) / 256), 256, 0, stream>>>(Wv, Wvb, (int)we);
    cvt_f32_bf16<<<(int)((we + 255) / 256), 256, 0, stream>>>(Wo, Wob, (int)we);

    // 2) QKV projections (V stored pre-transposed per head)
    dim3 ggrid(H_ / 64, NT_ / 128);          // (16, 64)
    gemm_bf16_kernel<<<ggrid, 128, 0, stream>>>(Xb, Wqb, Qb, NT_, H_, H_, 0);
    gemm_bf16_kernel<<<ggrid, 128, 0, stream>>>(Xb, Wkb, Kb, NT_, H_, H_, 0);
    gemm_bf16_kernel<<<ggrid, 128, 0, stream>>>(Xb, Wvb, Vb, NT_, H_, H_, 2);

    // 3) RoPE on Q and K
    const int rtotal = B_ * NH_ * S_ * 32;   // 4,194,304
    rope_kernel<<<rtotal / 256, 256, 0, stream>>>(Qb, rtotal);
    rope_kernel<<<rtotal / 256, 256, 0, stream>>>(Kb, rtotal);

    // 4) flash attention -> merged heads bf16 [NT, H]
    attn_kernel<<<B_ * NH_ * (S_ / 16), 32, 0, stream>>>(Qb, Kb, Vb, Ab);

    // 5) output projection -> f32 d_out
    gemm_bf16_kernel<<<ggrid, 128, 0, stream>>>(Ab, Wob, out, NT_, H_, H_, 1);
}
